// ablation2_51213190037999
// MI455X (gfx1250) — matrix operand order repaired and checked by reading, not compiled, not run on hardware
//
#include <hip/hip_runtime.h>
#include <hip/hip_bf16.h>
#include <stdint.h>

#define S_N 10000
#define R_N 16
#define F_N 128
#define H_N 4
#define B_N 2048
#define C_N 16

typedef _Float16 f16;
typedef __attribute__((ext_vector_type(16))) _Float16 v16h;
typedef __attribute__((ext_vector_type(8)))  float    v8f;

#define SB_STRIDE 136   // padded LDS row stride (halves) to stagger banks

__device__ __forceinline__ float lrelu(float x){ return x > 0.f ? x : 0.2f * x; }
__device__ __forceinline__ float elu1 (float x){ return x > 0.f ? x : __expf(x) - 1.f; }

// Async global->LDS 16B copy (CDNA5): LDS written directly, tracked by ASYNCcnt.
// lds_addr = low 32 bits of generic shared pointer (flat->LDS truncation, ISA 10.2).
__device__ __forceinline__ void async_b128(const f16* gsrc, f16* lds_dst){
  unsigned ld = (unsigned)(uintptr_t)lds_dst;
  unsigned long long ga = (unsigned long long)(uintptr_t)gsrc;
  asm volatile("global_load_async_to_lds_b128 %0, %1, off"
               :: "v"(ld), "v"(ga) : "memory");
}
__device__ __forceinline__ void wait_async0(){
  asm volatile("s_wait_asynccnt 0x0" ::: "memory");
}

// ---------------------------------------------------------------- conversions
__global__ void k_cvt(const float* __restrict__ in, f16* __restrict__ out, int n){
  int i = blockIdx.x * blockDim.x + threadIdx.x;
  if (i < n) out[i] = (f16)in[i];
}

__global__ void k_zero(float* __restrict__ p, int n){
  int i = blockIdx.x * blockDim.x + threadIdx.x;
  if (i < n) p[i] = 0.f;
}

// ------------------------------------------------- h1 = Rfeat@W1, e1 = h1·a_lo
__global__ void k_head_prep(const float* __restrict__ Rfeat, const float* __restrict__ W1,
                            const float* __restrict__ a, f16* __restrict__ h1h,
                            float* __restrict__ e1){
  int h = blockIdx.x, t = threadIdx.x;
  __shared__ float sh1[R_N * F_N];
  const float* W = W1 + (size_t)h * F_N * F_N;
  for (int e = t; e < R_N * F_N; e += blockDim.x){
    int r = e / F_N, f = e % F_N;
    float acc = 0.f;
    for (int k = 0; k < F_N; ++k) acc += Rfeat[r * F_N + k] * W[k * F_N + f];
    sh1[e] = acc;
    h1h[h * 32 * F_N + e] = (f16)acc;              // K rows 0..15
  }
  for (int e = t; e < R_N * F_N; e += blockDim.x)
    h1h[h * 32 * F_N + R_N * F_N + e] = (f16)0.f;  // zero-pad K rows 16..31
  __syncthreads();
  if (t < R_N){
    const float* alo = a + h * 2 * F_N;
    float acc = 0.f;
    for (int f = 0; f < F_N; ++f) acc += sh1[t * F_N + f] * alo[f];
    e1[h * R_N + t] = acc;
  }
}

// ---------- h2 = Sfeat @ W2 (WMMA; B tile async-staged in LDS, 8 waves/block)
__global__ void k_gemm_h2(const f16* __restrict__ A, const f16* __restrict__ W,
                          f16* __restrict__ h2h){
  __shared__ f16 sB[32 * SB_STRIDE];
  int t = threadIdx.x;
  int lane = t & 31, wv = t >> 5;
  int m = lane & 15, hf = lane >> 4;
  int s0 = blockIdx.x * 16, h = blockIdx.y;
  int n0 = wv * 16;
  const f16* Wh = W + (size_t)h * F_N * F_N;
  v8f c;
  #pragma unroll
  for (int i = 0; i < 8; ++i) c[i] = 0.f;
  for (int kc = 0; kc < 4; ++kc){
    __syncthreads();
    { // cooperative B-tile: 32 K-rows x 128 cols, two async 16B chunks per thread
      int k = t >> 3, col = (t & 7) * 16;
      const f16* sp = Wh + (size_t)(kc * 32 + k) * F_N + col;
      f16* dp = &sB[k * SB_STRIDE + col];
      async_b128(sp,     dp);
      async_b128(sp + 8, dp + 8);
    }
    wait_async0();
    __syncthreads();
    v16h av, bv;
    const f16* ar = A + (size_t)(s0 + m) * F_N + kc * 32 + hf * 8;
    #pragma unroll
    for (int e = 0; e < 8; ++e){ av[e] = ar[e]; av[e + 8] = ar[e + 16]; }
    #pragma unroll
    for (int e = 0; e < 16; ++e) bv[e] = sB[(8 * hf + (e < 8 ? e : e + 8)) * SB_STRIDE + n0 + m];
    c = __builtin_amdgcn_wmma_f32_16x16x32_f16(false, av, false, bv, (short)0, c, false, false);
  }
  f16* out = h2h + (size_t)h * S_N * F_N;
  #pragma unroll
  for (int i = 0; i < 8; ++i)
    out[(size_t)(s0 + i + 8 * hf) * F_N + n0 + m] = (f16)c[i];
}

// ------------------------------------- att_i = softmax_R(mask ? lrelu(e2+e1) )
__global__ void k_atti(const f16* __restrict__ h2h, const float* __restrict__ a,
                       const float* __restrict__ e1, const int* __restrict__ inter,
                       f16* __restrict__ attih){
  int idx = blockIdx.x * blockDim.x + threadIdx.x;
  if (idx >= H_N * S_N) return;
  int h = idx / S_N, s = idx % S_N;
  const f16* row = h2h + ((size_t)h * S_N + s) * F_N;
  const float* ahi = a + h * 2 * F_N + F_N;
  float e2 = 0.f;
  for (int f = 0; f < F_N; ++f) e2 += (float)row[f] * ahi[f];
  const int* mrow = inter + (size_t)s * R_N;
  float ev[R_N]; float mx = -3.0e38f; int any = 0;
  #pragma unroll
  for (int r = 0; r < R_N; ++r){
    float e12 = lrelu(e2 + e1[h * R_N + r]);
    ev[r] = e12;
    if (mrow[r] > 0){ any = 1; if (e12 > mx) mx = e12; }
  }
  f16* orow = attih + ((size_t)h * S_N + s) * R_N;
  if (!any){
    #pragma unroll
    for (int r = 0; r < R_N; ++r) orow[r] = (f16)(1.f / R_N);
    return;
  }
  float wv[R_N]; float sum = 0.f;
  #pragma unroll
  for (int r = 0; r < R_N; ++r){
    wv[r] = (mrow[r] > 0) ? __expf(ev[r] - mx) : 0.f;
    sum += wv[r];
  }
  float inv = 1.f / sum;
  #pragma unroll
  for (int r = 0; r < R_N; ++r) orow[r] = (f16)(wv[r] * inv);
}

// -------- att34[b,s] = city_m/deg3 + prov_m/deg4 (uniform 1/S if deg==0)
// Single pass over each gathered adjacency row; masks staged as LDS bitmaps.
__global__ void k_att34(const int* __restrict__ city, const int* __restrict__ prov,
                        const int* __restrict__ src, f16* __restrict__ att34){
  const int Wn = (S_N + 31) / 32;   // 313 words
  __shared__ unsigned sm3[320], sm4[320];
  __shared__ int sdeg[2];
  int b = blockIdx.x, t = threadIdx.x;
  if (t < 2) sdeg[t] = 0;
  __syncthreads();
  int sb = src[b];
  const int* c = city + (size_t)sb * S_N;
  const int* p = prov + (size_t)sb * S_N;
  int d3 = 0, d4 = 0;
  for (int w = t; w < Wn; w += blockDim.x){
    int base = w * 32;
    // stream-ahead hint for the next strip of both adjacency rows
    __builtin_prefetch(c + base + 32 * 256, 0, 0);
    __builtin_prefetch(p + base + 32 * 256, 0, 0);
    unsigned m3 = 0, m4 = 0;
    int lim = (S_N - base) < 32 ? (S_N - base) : 32;
    for (int j = 0; j < lim; ++j){
      if (c[base + j] > 0) m3 |= 1u << j;
      if (p[base + j] > 0) m4 |= 1u << j;
    }
    sm3[w] = m3; sm4[w] = m4;
    d3 += __popc(m3); d4 += __popc(m4);
  }
  atomicAdd(&sdeg[0], d3);
  atomicAdd(&sdeg[1], d4);
  __syncthreads();
  float r3 = sdeg[0] > 0 ? 1.f / (float)sdeg[0] : 0.f;
  float r4 = sdeg[1] > 0 ? 1.f / (float)sdeg[1] : 0.f;
  float u3 = sdeg[0] > 0 ? 0.f : 1.f / (float)S_N;   // all-masked softmax -> uniform
  float u4 = sdeg[1] > 0 ? 0.f : 1.f / (float)S_N;
  f16* orow = att34 + (size_t)b * S_N;
  for (int s = t; s < S_N; s += blockDim.x){
    float v3 = ((sm3[s >> 5] >> (s & 31)) & 1) ? r3 : 0.f;
    float v4 = ((sm4[s >> 5] >> (s & 31)) & 1) ? r4 : 0.f;
    orow[s] = (f16)(v3 + u3 + v4 + u4);
  }
}

// ------------------------------------------------------------- h2s = h2[src]
__global__ void k_gather(const f16* __restrict__ h2h, const int* __restrict__ src,
                         f16* __restrict__ h2s){
  int idx = blockIdx.x * blockDim.x + threadIdx.x;
  if (idx >= H_N * B_N * F_N) return;
  int f = idx % F_N; int hb = idx / F_N; int b = hb % B_N; int h = hb / B_N;
  h2s[idx] = h2h[((size_t)h * S_N + src[b]) * F_N + f];
}

// ---- upre = att34ᵀ@h2s + att_i@h1  (WMMA; async LDS-staged tiles, 8 waves)
__global__ void k_gemm_intra(const f16* __restrict__ att34, const f16* __restrict__ h2s,
                             const f16* __restrict__ attih, const f16* __restrict__ h1h,
                             float* __restrict__ upre){
  __shared__ f16 sA[32 * 16];
  __shared__ f16 sB[32 * SB_STRIDE];
  int t = threadIdx.x;
  int lane = t & 31, wv = t >> 5;
  int m = lane & 15, hf = lane >> 4;
  int s0 = blockIdx.x * 16, h = blockIdx.y;
  int n0 = wv * 16;
  const f16* Bh = h2s + (size_t)h * B_N * F_N;
  v8f c;
  #pragma unroll
  for (int i = 0; i < 8; ++i) c[i] = 0.f;
  for (int kc = 0; kc < B_N / 32; ++kc){
    __syncthreads();
    if (t < 64){  // A tile: rows k (=b), 16 s-columns contiguous in att34[b, s0..]
      int k = t >> 1, part = t & 1;
      async_b128(att34 + (size_t)(kc * 32 + k) * S_N + s0 + part * 8,
                 &sA[k * 16 + part * 8]);
    }
    { // B tile: 32 K-rows x 128 cols of h2s
      int k = t >> 3, col = (t & 7) * 16;
      const f16* sp = Bh + (size_t)(kc * 32 + k) * F_N + col;
      f16* dp = &sB[k * SB_STRIDE + col];
      async_b128(sp,     dp);
      async_b128(sp + 8, dp + 8);
    }
    wait_async0();
    __syncthreads();
    v16h av, bv;
    #pragma unroll
    for (int e = 0; e < 16; ++e){
      int k0 = 8 * hf + (e < 8 ? e : e + 8);
      av[e] = sA[k0 * 16 + m];
    }
    #pragma unroll
    for (int e = 0; e < 16; ++e)
      bv[e] = sB[(8 * hf + (e < 8 ? e : e + 8)) * SB_STRIDE + n0 + m];
    c = __builtin_amdgcn_wmma_f32_16x16x32_f16(false, av, false, bv, (short)0, c, false, false);
  }
  { // fused epilogue: InterRC = att_i @ h1, K=16 zero-padded to 32
    v16h av, bv;
    const f16* arow = attih + (size_t)h * S_N * R_N;
    #pragma unroll
    for (int e = 0; e < 16; ++e){
      int k0 = 8 * hf + (e < 8 ? e : e + 8);
      av[e] = (k0 < R_N) ? arow[(size_t)(s0 + m) * R_N + k0] : (f16)0.f;
    }
    const f16* hb = h1h + h * 32 * F_N;            // rows 16..31 are zero
    #pragma unroll
    for (int e = 0; e < 16; ++e)
      bv[e] = hb[(8 * hf + (e < 8 ? e : e + 8)) * F_N + n0 + m];
    c = __builtin_amdgcn_wmma_f32_16x16x32_f16(false, av, false, bv, (short)0, c, false, false);
  }
  float* out = upre + (size_t)h * S_N * F_N;
  #pragma unroll
  for (int i = 0; i < 8; ++i)
    out[(size_t)(s0 + i + 8 * hf) * F_N + n0 + m] = c[i];
}

// ------------------------------------------------ BN2 batch stats over S axis
__global__ void k_stats(const float* __restrict__ upre, float* __restrict__ sum,
                        float* __restrict__ sq){
  int h = blockIdx.x, chunk = blockIdx.y, f = threadIdx.x;
  const float* base = upre + (size_t)h * S_N * F_N;
  int s0 = chunk * 200;
  float a = 0.f, b = 0.f;
  for (int s = s0; s < s0 + 200; ++s){
    float v = base[(size_t)s * F_N + f];
    a += v; b += v * v;
  }
  atomicAdd(&sum[h * F_N + f], a);
  atomicAdd(&sq [h * F_N + f], b);
}

// ------------------------------------------------- u = lrelu(BN2(upre)) -> f16
__global__ void k_u(const float* __restrict__ upre, const float* __restrict__ sum,
                    const float* __restrict__ sq, const float* __restrict__ g,
                    const float* __restrict__ bb, f16* __restrict__ uh){
  size_t idx = (size_t)blockIdx.x * blockDim.x + threadIdx.x;
  if (idx >= (size_t)H_N * S_N * F_N) return;
  int f = (int)(idx % F_N);
  int h = (int)(idx / ((size_t)S_N * F_N));
  float mean = sum[h * F_N + f] * (1.f / S_N);
  float var  = sq [h * F_N + f] * (1.f / S_N) - mean * mean;
  float x = upre[idx];
  float y = (x - mean) * rsqrtf(var + 1e-5f) * g[h * F_N + f] + bb[h * F_N + f];
  uh[idx] = (f16)lrelu(y);
}

// ---- vpre += att_iᵀ @ h2 (WMMA; async LDS tiles, K split over 25 blocks/head)
#define V_KSEG 13
__global__ void k_gemm_v(const f16* __restrict__ attih, const f16* __restrict__ h2h,
                         float* __restrict__ vpre){
  __shared__ f16 sA[32 * 16];
  __shared__ f16 sB[32 * SB_STRIDE];
  int t = threadIdx.x;
  int lane = t & 31, wv = t >> 5;
  int m = lane & 15, hf = lane >> 4;
  int h = blockIdx.x, seg = blockIdx.y;
  int n0 = wv * 16;
  const f16* Ah = attih + (size_t)h * S_N * R_N;
  const f16* Bh = h2h   + (size_t)h * S_N * F_N;
  const int KC = (S_N + 31) / 32;                  // 313
  int kcB = seg * V_KSEG;
  int kcE = kcB + V_KSEG < KC ? kcB + V_KSEG : KC;
  v8f c;
  #pragma unroll
  for (int i = 0; i < 8; ++i) c[i] = 0.f;
  for (int kc = kcB; kc < kcE; ++kc){
    __syncthreads();
    if (t < 64){  // A tile: row s -> 16 r-columns contiguous in att_i[s, :]
      int k = t >> 1, part = t & 1;
      int s = kc * 32 + k;
      if (s < S_N)
        async_b128(Ah + (size_t)s * R_N + part * 8, &sA[k * 16 + part * 8]);
      else
        *(uint4*)&sA[k * 16 + part * 8] = make_uint4(0u, 0u, 0u, 0u);
    }
    { // B tile: 32 s-rows x 128 cols of h2 (zero past S)
      int k = t >> 3, col = (t & 7) * 16;
      int s = kc * 32 + k;
      f16* dp = &sB[k * SB_STRIDE + col];
      if (s < S_N){
        const f16* sp = Bh + (size_t)s * F_N + col;
        async_b128(sp,     dp);
        async_b128(sp + 8, dp + 8);
      } else {
        uint4* d4 = (uint4*)dp;
        d4[0] = make_uint4(0u, 0u, 0u, 0u);
        d4[1] = make_uint4(0u, 0u, 0u, 0u);
      }
    }
    wait_async0();
    __syncthreads();
    v16h av, bv;
    #pragma unroll
    for (int e = 0; e < 16; ++e){
      int k0 = 8 * hf + (e < 8 ? e : e + 8);
      av[e] = sA[k0 * 16 + m];
    }
    #pragma unroll
    for (int e = 0; e < 16; ++e)
      bv[e] = sB[(8 * hf + (e < 8 ? e : e + 8)) * SB_STRIDE + n0 + m];
    c = __builtin_amdgcn_wmma_f32_16x16x32_f16(false, av, false, bv, (short)0, c, false, false);
  }
  #pragma unroll
  for (int i = 0; i < 8; ++i)
    atomicAdd(&vpre[h * R_N * F_N + (i + 8 * hf) * F_N + n0 + m], c[i]);
}

// ----------------------------------------------- v = lrelu(BN1(vpre)) -> f16
__global__ void k_v_bn(const float* __restrict__ vpre, const float* __restrict__ g,
                       const float* __restrict__ bb, f16* __restrict__ vh){
  int h = blockIdx.x, f = threadIdx.x;
  const float* col = vpre + h * R_N * F_N + f;
  float mean = 0.f;
  for (int r = 0; r < R_N; ++r) mean += col[r * F_N];
  mean *= (1.f / R_N);
  float var = 0.f;
  for (int r = 0; r < R_N; ++r){ float d = col[r * F_N] - mean; var += d * d; }
  var *= (1.f / R_N);
  float is = rsqrtf(var + 1e-5f);
  float gg = g[h * F_N + f], bbv = bb[h * F_N + f];
  for (int r = 0; r < R_N; ++r){
    float y = (col[r * F_N] - mean) * is * gg + bbv;
    vh[h * R_N * F_N + r * F_N + f] = (f16)lrelu(y);
  }
}

// ----------------------------------- x[s, h*16+r] = elu(u @ vᵀ) (WMMA, K=128)
__global__ void k_gemm_hx(const f16* __restrict__ uh, const f16* __restrict__ vh,
                          float* __restrict__ x){
  int lane = threadIdx.x; int m = lane & 15, hf = lane >> 4;
  int s0 = blockIdx.x * 16, h = blockIdx.y;
  const f16* Ah = uh + (size_t)h * S_N * F_N;
  const f16* Vh = vh + h * R_N * F_N;
  v8f c;
  #pragma unroll
  for (int i = 0; i < 8; ++i) c[i] = 0.f;
  #pragma unroll
  for (int kc = 0; kc < 4; ++kc){
    v16h av, bv;
    const f16* ar = Ah + (size_t)(s0 + m) * F_N + kc * 32 + hf * 8;
    #pragma unroll
    for (int e = 0; e < 8; ++e){ av[e] = ar[e]; av[e + 8] = ar[e + 16]; }
    const f16* br = Vh + m * F_N + kc * 32 + 8 * hf;   // B = vᵀ, table K order (8-runs)
    #pragma unroll
    for (int e = 0; e < 16; ++e) bv[e] = br[e < 8 ? e : e + 8];
    c = __builtin_amdgcn_wmma_f32_16x16x32_f16(false, av, false, bv, (short)0, c, false, false);
  }
  #pragma unroll
  for (int i = 0; i < 8; ++i)
    x[(size_t)(s0 + i + 8 * hf) * (H_N * R_N) + h * R_N + m] = elu1(c[i]);
}

// --------- output GAT layer: degree-uniform attention + double elu + logsoftmax
__global__ void k_out(const float* __restrict__ x, const float* __restrict__ outW,
                      const int* __restrict__ inter, float* __restrict__ out){
  __shared__ float sW[64 * C_N];
  int t = threadIdx.x;
  for (int i = t; i < 64 * C_N; i += blockDim.x) sW[i] = outW[i];
  __syncthreads();
  int s = blockIdx.x * blockDim.x + t;
  if (s >= S_N) return;
  const float* xr = x + (size_t)s * 64;
  float hc[C_N];
  #pragma unroll
  for (int c0 = 0; c0 < C_N; ++c0) hc[c0] = 0.f;
  for (int k = 0; k < 64; ++k){
    float xv = xr[k];
    #pragma unroll
    for (int c0 = 0; c0 < C_N; ++c0) hc[c0] += xv * sW[k * C_N + c0];
  }
  const int* mrow = inter + (size_t)s * R_N;
  int deg = 0;
  #pragma unroll
  for (int r = 0; r < R_N; ++r) deg += (mrow[r] > 0) ? 1 : 0;
  float y[C_N];
  #pragma unroll
  for (int c0 = 0; c0 < C_N; ++c0){
    float att = (deg > 0) ? ((mrow[c0] > 0) ? 1.f / (float)deg : 0.f) : 1.f / (float)R_N;
    y[c0] = elu1(elu1(att * hc[c0]));
  }
  float mx = -3.0e38f;
  #pragma unroll
  for (int c0 = 0; c0 < C_N; ++c0) mx = y[c0] > mx ? y[c0] : mx;
  float sum = 0.f;
  #pragma unroll
  for (int c0 = 0; c0 < C_N; ++c0) sum += __expf(y[c0] - mx);
  float lse = mx + __logf(sum);
  #pragma unroll
  for (int c0 = 0; c0 < C_N; ++c0) out[(size_t)s * C_N + c0] = y[c0] - lse;
}

// ============================================================================
extern "C" void kernel_launch(void* const* d_in, const int* in_sizes, int n_in,
                              void* d_out, int out_size, void* d_ws, size_t ws_size,
                              hipStream_t stream) {
  (void)in_sizes; (void)n_in; (void)out_size; (void)ws_size;
  const int*   inter = (const int*)d_in[0];
  const int*   city  = (const int*)d_in[1];
  const int*   prov  = (const int*)d_in[2];
  const int*   src   = (const int*)d_in[3];
  const float* Sfeat = (const float*)d_in[4];
  const float* Rfeat = (const float*)d_in[5];
  const float* W1    = (const float*)d_in[6];
  const float* W2    = (const float*)d_in[7];
  const float* a_    = (const float*)d_in[8];
  // d_in[9], d_in[10] (a3, a4) are provably unused: constant-score softmax.
  const float* bn1g  = (const float*)d_in[11];
  const float* bn1b  = (const float*)d_in[12];
  const float* bn2g  = (const float*)d_in[13];
  const float* bn2b  = (const float*)d_in[14];
  const float* outW  = (const float*)d_in[15];
  float* out = (float*)d_out;

  char* w = (char*)d_ws;
  auto alloc = [&](size_t bytes) -> char* {
    char* p = w; w += (bytes + 255) & ~(size_t)255; return p;
  };
  f16*   sfeat_h = (f16*)  alloc((size_t)S_N * F_N * 2);
  f16*   w2_h    = (f16*)  alloc((size_t)H_N * F_N * F_N * 2);
  f16*   h1_h    = (f16*)  alloc((size_t)H_N * 32 * F_N * 2);
  float* e1_f    = (float*)alloc((size_t)H_N * R_N * 4);
  f16*   h2_h    = (f16*)  alloc((size_t)H_N * S_N * F_N * 2);
  f16*   atti_h  = (f16*)  alloc((size_t)H_N * S_N * R_N * 2);
  f16*   att34_h = (f16*)  alloc((size_t)B_N * S_N * 2);
  f16*   h2s_h   = (f16*)  alloc((size_t)H_N * B_N * F_N * 2);
  float* upre    = (float*)alloc((size_t)H_N * S_N * F_N * 4);
  float* ssum    = (float*)alloc((size_t)H_N * F_N * 4);
  float* ssq     = (float*)alloc((size_t)H_N * F_N * 4);
  f16*   u_h     = (f16*)  alloc((size_t)H_N * S_N * F_N * 2);
  float* vpre    = (float*)alloc((size_t)H_N * R_N * F_N * 4);
  f16*   v_h     = (f16*)  alloc((size_t)H_N * R_N * F_N * 2);
  float* x_f     = (float*)alloc((size_t)S_N * H_N * R_N * 4);

  // 1) f32 -> f16 conversions for the WMMA operands
  k_cvt<<<(S_N * F_N + 255) / 256, 256, 0, stream>>>(Sfeat, sfeat_h, S_N * F_N);
  k_cvt<<<(H_N * F_N * F_N + 255) / 256, 256, 0, stream>>>(W2, w2_h, H_N * F_N * F_N);

  // 2) tiny per-head prep: h1 (K-padded f16) + e1
  k_head_prep<<<H_N, 256, 0, stream>>>(Rfeat, W1, a_, h1_h, e1_f);

  // 3) h2 = Sfeat @ W2  (WMMA, async LDS-staged B, 8 waves/block)
  k_gemm_h2<<<dim3(S_N / 16, H_N), 256, 0, stream>>>(sfeat_h, w2_h, h2_h);

  // 4) att_i masked softmax over R
  k_atti<<<(H_N * S_N + 255) / 256, 256, 0, stream>>>(h2_h, a_, e1_f, inter, atti_h);

  // 5) degree-uniform att3+att4 (single adjacency pass + prefetch streaming)
  k_att34<<<B_N, 256, 0, stream>>>(city, prov, src, att34_h);

  // 6) gather h2s = h2[src]
  k_gather<<<(H_N * B_N * F_N + 255) / 256, 256, 0, stream>>>(h2_h, src, h2s_h);

  // 7) upre = att34ᵀ@h2s + att_i@h1  (WMMA, async LDS tiles, fused epilogue)
  k_gemm_intra<<<dim3(S_N / 16, H_N), 256, 0, stream>>>(att34_h, h2s_h,
                                                        atti_h, h1_h, upre);

  // 8) BN2 stats + apply, u in f16
  k_zero<<<(2 * H_N * F_N + 255) / 256, 256, 0, stream>>>(ssum, 2 * H_N * F_N); // ssum+ssq contiguous
  k_stats<<<dim3(H_N, 50), 128, 0, stream>>>(upre, ssum, ssq);
  k_u<<<(int)(((size_t)H_N * S_N * F_N + 255) / 256), 256, 0, stream>>>(upre, ssum, ssq,
                                                                        bn2g, bn2b, u_h);

  // 9) vpre = att_iᵀ @ h2 (WMMA, K-split + f32 atomic reduce), then BN1 over R
  k_zero<<<(H_N * R_N * F_N + 255) / 256, 256, 0, stream>>>(vpre, H_N * R_N * F_N);
  k_gemm_v<<<dim3(H_N, 25), 256, 0, stream>>>(atti_h, h2_h, vpre);
  k_v_bn<<<H_N, 128, 0, stream>>>(vpre, bn1g, bn1b, v_h);

  // 10) x = elu(u @ vᵀ) per head (WMMA)
  k_gemm_hx<<<dim3(S_N / 16, H_N), 32, 0, stream>>>(u_h, v_h, x_f);

  // 11) output layer (degree-uniform attention, double elu, log_softmax)
  k_out<<<(S_N + 127) / 128, 128, 0, stream>>>(x_f, outW, inter, out);
}
